// Indexer_2027224563741
// MI455X (gfx1250) — matrix operand order repaired and checked by reading, not compiled, not run on hardware
//
#include <hip/hip_runtime.h>

// ---------------------------------------------------------------------------
// Types for CDNA5 WMMA (wave32): v_wmma_f32_16x16x32_bf16
// ---------------------------------------------------------------------------
typedef __bf16 bf16_t;
typedef __attribute__((ext_vector_type(8)))  __bf16 v8bf;
typedef __attribute__((ext_vector_type(16))) __bf16 v16bf;
typedef __attribute__((ext_vector_type(8)))  float  v8f;
typedef int v4i_ __attribute__((vector_size(16)));  // builtin's pointee type

static __device__ __forceinline__ v16bf cat16(v8bf lo, v8bf hi) {
    return __builtin_shufflevector(lo, hi, 0, 1, 2, 3, 4, 5, 6, 7,
                                           8, 9, 10, 11, 12, 13, 14, 15);
}

static __device__ __forceinline__ v8f zero8f() {
    v8f z = {0.f, 0.f, 0.f, 0.f, 0.f, 0.f, 0.f, 0.f};
    return z;
}

// ---------------------------------------------------------------------------
// Async global -> LDS staging (gfx1250 GLOBAL_LOAD_ASYNC_TO_LDS_B128,
// tracked by ASYNCcnt).  Fallback: plain copy through VGPRs (DScnt-fenced
// by __syncthreads()).
// ---------------------------------------------------------------------------
#if __has_builtin(__builtin_amdgcn_global_load_async_to_lds_b128)
#define HAS_ASYNC_LDS 1
#else
#define HAS_ASYNC_LDS 0
#endif

static __device__ __forceinline__ void copy16_to_lds(bf16_t* lds_dst,
                                                     const bf16_t* gsrc) {
#if HAS_ASYNC_LDS
    __builtin_amdgcn_global_load_async_to_lds_b128((v4i_*)gsrc, (v4i_*)lds_dst,
                                                   0, 0);
#else
    *(v8bf*)lds_dst = *(const v8bf*)gsrc;
#endif
}

static __device__ __forceinline__ void wait_async_copies() {
#if HAS_ASYNC_LDS
#if __has_builtin(__builtin_amdgcn_s_wait_asynccnt)
    __builtin_amdgcn_s_wait_asynccnt(0);
#else
    asm volatile("s_wait_asynccnt 0x0" ::: "memory");
#endif
#endif
}

// batch start offset for token t given seq_lens[0..nb)
static __device__ __forceinline__ int batch_start(const int* seq_lens, int nb, int t) {
    int off = 0;
    for (int b = 0; b < nb; ++b) {
        int L = seq_lens[b];
        if (t < off + L) return off;
        off += L;
    }
    return off;
}

#define LOG2_ROPE_OVER_HALF 0.41524101186f  // log2(10000)/32

// ---------------------------------------------------------------------------
// 1) elementwise f32 -> bf16
// ---------------------------------------------------------------------------
__global__ __launch_bounds__(256) void cvt_f32_bf16(const float* __restrict__ s,
                                                    bf16_t* __restrict__ d, size_t n) {
    size_t i = (size_t)blockIdx.x * blockDim.x + threadIdx.x;
    size_t stride = (size_t)gridDim.x * blockDim.x;
    for (; i < n; i += stride) d[i] = (bf16_t)s[i];
}

// ---------------------------------------------------------------------------
// 2) q = qr @ wq_b.T with fused RoPE epilogue, bf16 output.
//    Block: 8 waves, 128 tokens x 128 cols (one head).  The 128x32 bf16
//    B k-slice (8 KB) is async-staged into LDS (double-buffered) and shared
//    by all 8 waves; each wave runs a 16x128 tile (8 WMMA accumulators).
// ---------------------------------------------------------------------------
__global__ __launch_bounds__(256)
void gemm_q_rope(const bf16_t* __restrict__ A, const bf16_t* __restrict__ B,
                 bf16_t* __restrict__ Q, const int* __restrict__ seq_lens,
                 int nb, int T, int K) {
    __shared__ bf16_t Bs[2][128 * 32];  // 2 x 8 KB

    const int lane = threadIdx.x & 31;
    const int lanelo = lane & 15;
    const int hi = lane >> 4;
    const int w = threadIdx.x >> 5;  // wave in block: M sub-tile

    const int NHEADS = 64;
    const int mtBase = (blockIdx.x / NHEADS) * 128;
    const int h = blockIdx.x % NHEADS;
    if (mtBase >= T) return;

    const int ka = hi ? 16 : 0;  // A fragment K offset (contiguous, matches B)
    const int kb = hi ? 16 : 0;  // B fragment K offset (contiguous, matches A)

    size_t arow = (size_t)(mtBase + w * 16 + lanelo) * K;

    v8f acc[8];
#pragma unroll
    for (int j = 0; j < 8; ++j) acc[j] = zero8f();

    // cooperative fill of one 128x32 B k-slice: 512 units of 16 bytes
    auto fill_B = [&](int k0, int buf) {
#pragma unroll
        for (int u = threadIdx.x; u < 512; u += 256) {
            int row = u >> 2;
            int seg = (u & 3) * 8;
            copy16_to_lds(&Bs[buf][row * 32 + seg],
                          B + (size_t)(h * 128 + row) * K + k0 + seg);
        }
    };

    const int nsteps = K >> 5;
    fill_B(0, 0);
    wait_async_copies();
    __syncthreads();

    for (int i = 0; i < nsteps; ++i) {
        int cur = i & 1;
        if (i + 1 < nsteps) fill_B((i + 1) << 5, cur ^ 1);

        int k0 = i << 5;
        v16bf av = cat16(*(const v8bf*)(A + arow + k0 + ka),
                         *(const v8bf*)(A + arow + k0 + ka + 8));
#pragma unroll
        for (int j = 0; j < 8; ++j) {
            const bf16_t* bp = &Bs[cur][(j * 16 + lanelo) * 32 + kb];
            v16bf bvf = cat16(*(const v8bf*)bp, *(const v8bf*)(bp + 8));
            acc[j] = __builtin_amdgcn_wmma_f32_16x16x32_bf16(
                false, av, false, bvf, (short)0, acc[j], false, false);
        }
        wait_async_copies();
        __syncthreads();
    }

    // Epilogue: RoPE on head dims [0,64): pairs (d, d+32); dims [64,128) pass.
    // D layout: element r of acc[j] -> row m = r + hi*8, col d = j*16 + lanelo.
#pragma unroll
    for (int r = 0; r < 8; ++r) {
        int t = mtBase + w * 16 + hi * 8 + r;
        float pos = (float)(t - batch_start(seq_lens, nb, t));
        bf16_t* qout = Q + (size_t)t * 8192 + h * 128;
#pragma unroll
        for (int j = 0; j < 2; ++j) {
            int i = j * 16 + lanelo;  // 0..31
            float inv = exp2f(-(float)i * LOG2_ROPE_OVER_HALF);
            float ang = pos * inv;
            float c = __cosf(ang), s = __sinf(ang);
            float x1 = acc[j][r];
            float x2 = acc[j + 2][r];
            qout[i] = (bf16_t)(x1 * c - x2 * s);
            qout[i + 32] = (bf16_t)(x1 * s + x2 * c);
        }
#pragma unroll
        for (int j = 4; j < 8; ++j) qout[j * 16 + lanelo] = (bf16_t)acc[j][r];
    }
}

// ---------------------------------------------------------------------------
// 3) Generic C[m,n] = sum_k A[m,k] * B[n,k]; A: MxK bf16, B: NxK bf16,
//    C: MxN f32.  Wave tile 16 x 64.  Used only for the small GEMMs
//    (k projection N=128, w projection N=64) which are bandwidth-trivial.
// ---------------------------------------------------------------------------
__global__ __launch_bounds__(256)
void gemm_bf16_nt(const bf16_t* __restrict__ A, const bf16_t* __restrict__ B,
                  float* __restrict__ C, int M, int N, int K) {
    const int lane = threadIdx.x & 31;
    const int lanelo = lane & 15;
    const int hi = lane >> 4;
    int wave = blockIdx.x * (blockDim.x >> 5) + (threadIdx.x >> 5);
    int tilesN = N >> 6;
    int mt = wave / tilesN;
    int nt = wave - mt * tilesN;
    if (mt * 16 >= M) return;

    const int ka = hi ? 16 : 0;
    const int kb = hi ? 16 : 0;
    size_t arow = (size_t)(mt * 16 + lanelo) * K;

    v8f acc[4];
#pragma unroll
    for (int j = 0; j < 4; ++j) acc[j] = zero8f();

    for (int k0 = 0; k0 < K; k0 += 32) {
        v16bf av = cat16(*(const v8bf*)(A + arow + k0 + ka),
                         *(const v8bf*)(A + arow + k0 + ka + 8));
#pragma unroll
        for (int j = 0; j < 4; ++j) {
            size_t brow = (size_t)(nt * 64 + j * 16 + lanelo) * K;
            v16bf bvf = cat16(*(const v8bf*)(B + brow + k0 + kb),
                              *(const v8bf*)(B + brow + k0 + kb + 8));
            acc[j] = __builtin_amdgcn_wmma_f32_16x16x32_bf16(
                false, av, false, bvf, (short)0, acc[j], false, false);
        }
    }

    int rbase = mt * 16 + hi * 8;
#pragma unroll
    for (int j = 0; j < 4; ++j) {
        int col = nt * 64 + j * 16 + lanelo;
#pragma unroll
        for (int r = 0; r < 8; ++r)
            C[(size_t)(rbase + r) * N + col] = acc[j][r];
    }
}

// ---------------------------------------------------------------------------
// 4) LayerNorm(k_raw) * w + b, then RoPE on dims [0,64); bf16 output.
// ---------------------------------------------------------------------------
__global__ __launch_bounds__(128)
void ln_rope_k(const float* __restrict__ kraw, const float* __restrict__ gw,
               const float* __restrict__ gb, const int* __restrict__ seq_lens,
               int nb, bf16_t* __restrict__ kout, int T) {
    __shared__ float red[128];
    __shared__ float row[128];
    int t = blockIdx.x;
    int d = threadIdx.x;
    float x = kraw[(size_t)t * 128 + d];
    red[d] = x;
    __syncthreads();
    for (int s = 64; s > 0; s >>= 1) {
        if (d < s) red[d] += red[d + s];
        __syncthreads();
    }
    float mu = red[0] * (1.0f / 128.0f);
    __syncthreads();
    float c = x - mu;
    red[d] = c * c;
    __syncthreads();
    for (int s = 64; s > 0; s >>= 1) {
        if (d < s) red[d] += red[d + s];
        __syncthreads();
    }
    float var = red[0] * (1.0f / 128.0f);
    float y = c * rsqrtf(var + 1e-6f) * gw[d] + gb[d];
    row[d] = y;
    __syncthreads();

    float pos = (float)(t - batch_start(seq_lens, nb, t));
    float out;
    if (d < 32) {
        float ang = pos * exp2f(-(float)d * LOG2_ROPE_OVER_HALF);
        out = row[d] * __cosf(ang) - row[d + 32] * __sinf(ang);
    } else if (d < 64) {
        int i = d - 32;
        float ang = pos * exp2f(-(float)i * LOG2_ROPE_OVER_HALF);
        out = row[i] * __sinf(ang) + row[d] * __cosf(ang);
    } else {
        out = y;
    }
    kout[(size_t)t * 128 + d] = (bf16_t)out;
}

// ---------------------------------------------------------------------------
// 5) logits[t,s] = mask( scale * sum_h w[t,h]*relu(q[t,h,:]·k[s,:]) )
//    Block: 8 waves share one t-tile (16 tokens); waves take 8 consecutive
//    s-tiles.  The shared 16x128 q head-slice (4 KB) is async-staged in LDS
//    per head (double-buffered over the 64-head loop); the per-wave k-tile
//    B fragments live in registers; the 16x64 w tile sits in LDS.
// ---------------------------------------------------------------------------
__global__ __launch_bounds__(256)
void logits_kernel(const bf16_t* __restrict__ q, const bf16_t* __restrict__ kmat,
                   const float* __restrict__ w, const int* __restrict__ seq_lens,
                   int nb, float* __restrict__ out, int T) {
    __shared__ bf16_t Qs[2][16 * 128];  // 2 x 4 KB
    __shared__ float Ws[16 * 64];       // 4 KB

    const int lane = threadIdx.x & 31;
    const int lanelo = lane & 15;
    const int hi = lane >> 4;
    const int wv = threadIdx.x >> 5;

    int tilesS8 = (T >> 4) >> 3;              // s-tile groups per row
    int tm = blockIdx.x / tilesS8;            // t tile
    int tn = (blockIdx.x % tilesS8) * 8 + wv; // s tile for this wave
    if (tm * 16 >= T) return;

    const int ka = hi ? 16 : 0;
    const int kb = hi ? 16 : 0;

    // w tile -> LDS (rows tm*16..+15, all 64 heads: contiguous 4 KB)
#pragma unroll
    for (int i = threadIdx.x; i < 16 * 64; i += 256)
        Ws[i] = w[(size_t)tm * 16 * 64 + i];

    // B fragments: k rows s = tn*16 + lanelo, K = 128 (4 chunks of 32)
    size_t brow = (size_t)(tn * 16 + lanelo) * 128;
    v16bf bv[4];
#pragma unroll
    for (int j = 0; j < 4; ++j)
        bv[j] = cat16(*(const v8bf*)(kmat + brow + j * 32 + kb),
                      *(const v8bf*)(kmat + brow + j * 32 + kb + 8));

    // cooperative fill of the 16x128 q slice for head h: 256 units of 16 B
    auto fill_Q = [&](int h, int buf) {
        int u = threadIdx.x;
        int row = u >> 4;
        int seg = (u & 15) * 8;
        copy16_to_lds(&Qs[buf][row * 128 + seg],
                      q + (size_t)(tm * 16 + row) * 8192 + h * 128 + seg);
    };

    float acc[8] = {0.f, 0.f, 0.f, 0.f, 0.f, 0.f, 0.f, 0.f};

    fill_Q(0, 0);
    wait_async_copies();
    __syncthreads();

    for (int h = 0; h < 64; ++h) {
        int cur = h & 1;
        if (h + 1 < 64) fill_Q(h + 1, cur ^ 1);

        v8f sc = zero8f();
#pragma unroll
        for (int j = 0; j < 4; ++j) {
            const bf16_t* ap = &Qs[cur][lanelo * 128 + j * 32 + ka];
            v16bf av = cat16(*(const v8bf*)ap, *(const v8bf*)(ap + 8));
            sc = __builtin_amdgcn_wmma_f32_16x16x32_bf16(
                false, av, false, bv[j], (short)0, sc, false, false);
        }
#pragma unroll
        for (int r = 0; r < 8; ++r) {
            float v = sc[r];
            v = v > 0.f ? v : 0.f;  // relu
            acc[r] += Ws[(hi * 8 + r) * 64 + h] * v;
        }
        wait_async_copies();
        __syncthreads();
    }

    const float scale = 0.08838834764831845f;  // 128^-0.5
    int scol = tn * 16 + lanelo;
#pragma unroll
    for (int r = 0; r < 8; ++r) {
        int t = tm * 16 + hi * 8 + r;
        int start = batch_start(seq_lens, nb, t);
        bool valid = (scol >= start) && (scol <= t);
        out[(size_t)t * T + scol] = valid ? acc[r] * scale : -1e30f;
    }
}

// ---------------------------------------------------------------------------
// 6) Per-row top-k (k=512) over T via 2048-wide bitonic sort in LDS,
//    descending by value, ascending index on ties (matches lax.top_k).
// ---------------------------------------------------------------------------
__global__ __launch_bounds__(256)
void topk_kernel(const float* __restrict__ logits, int T,
                 float* __restrict__ outv, int* __restrict__ outi) {
    __shared__ float sv[2048];
    __shared__ int si[2048];
    int t = blockIdx.x;
    for (int i = threadIdx.x; i < 2048; i += blockDim.x) {
        if (i < T) {
            sv[i] = logits[(size_t)t * T + i];
            si[i] = i;
        } else {
            sv[i] = -__builtin_inff();
            si[i] = 0x7fffffff;
        }
    }
    __syncthreads();
    for (int kk = 2; kk <= 2048; kk <<= 1) {
        for (int j = kk >> 1; j > 0; j >>= 1) {
            for (int i = threadIdx.x; i < 2048; i += blockDim.x) {
                int ixj = i ^ j;
                if (ixj > i) {
                    bool up = ((i & kk) == 0);  // descending run
                    float a = sv[i], b = sv[ixj];
                    int ai = si[i], bi = si[ixj];
                    bool a_less = (a < b) || (a == b && ai > bi);
                    if (up == a_less) {
                        sv[i] = b; sv[ixj] = a;
                        si[i] = bi; si[ixj] = ai;
                    }
                }
            }
            __syncthreads();
        }
    }
    for (int i = threadIdx.x; i < 512; i += blockDim.x) {
        outv[(size_t)t * 512 + i] = sv[i];
        outi[(size_t)t * 512 + i] = si[i];
    }
}

// ---------------------------------------------------------------------------
// Host launch
// ---------------------------------------------------------------------------
extern "C" void kernel_launch(void* const* d_in, const int* in_sizes, int n_in,
                              void* d_out, int out_size, void* d_ws, size_t ws_size,
                              hipStream_t stream) {
    const float* qr      = (const float*)d_in[0];
    const float* hidden  = (const float*)d_in[1];
    const float* wq_b    = (const float*)d_in[2];
    const float* wk      = (const float*)d_in[3];
    const float* knw     = (const float*)d_in[4];
    const float* knb     = (const float*)d_in[5];
    const float* wproj   = (const float*)d_in[6];
    const int* seq_lens  = (const int*)d_in[7];
    const int nb = in_sizes[7];

    const int QLORA = 1536, HID = 7168, NQ = 8192, HD = 128, NH = 64;
    const int T = in_sizes[0] / QLORA;

    // carve workspace (256B aligned chunks)
    char* p = (char*)d_ws;
    auto carve = [&](size_t bytes) -> void* {
        void* r = (void*)p;
        p += (bytes + 255) & ~(size_t)255;
        return r;
    };
    bf16_t* qr_b  = (bf16_t*)carve((size_t)T * QLORA * 2);
    bf16_t* wqb_b = (bf16_t*)carve((size_t)NQ * QLORA * 2);
    bf16_t* hid_b = (bf16_t*)carve((size_t)T * HID * 2);
    bf16_t* wk_b  = (bf16_t*)carve((size_t)HD * HID * 2);
    bf16_t* wp_b  = (bf16_t*)carve((size_t)NH * HID * 2);
    bf16_t* q_b   = (bf16_t*)carve((size_t)T * NQ * 2);
    float* kraw   = (float*)carve((size_t)T * HD * 4);
    bf16_t* k_b   = (bf16_t*)carve((size_t)T * HD * 2);
    float* w_f    = (float*)carve((size_t)T * NH * 4);
    float* logits = (float*)carve((size_t)T * (size_t)T * 4);

    auto cvt = [&](const float* s, bf16_t* d, size_t n) {
        int blocks = (int)((n + 255) / 256);
        if (blocks > 8192) blocks = 8192;
        cvt_f32_bf16<<<blocks, 256, 0, stream>>>(s, d, n);
    };
    cvt(qr, qr_b, (size_t)T * QLORA);
    cvt(wq_b, wqb_b, (size_t)NQ * QLORA);
    cvt(hidden, hid_b, (size_t)T * HID);
    cvt(wk, wk_b, (size_t)HD * HID);
    cvt(wproj, wp_b, (size_t)NH * HID);

    // q projection + RoPE: block = 128 tokens x 1 head, LDS-staged B
    {
        int blocks = (T / 128) * 64;
        gemm_q_rope<<<blocks, 256, 0, stream>>>(qr_b, wqb_b, q_b,
                                                seq_lens, nb, T, QLORA);
    }
    // k_raw = hidden @ wk.T   (M=T, N=128, K=7168)
    {
        int waves = (T / 16) * (HD / 64);
        gemm_bf16_nt<<<(waves + 7) / 8, 256, 0, stream>>>(hid_b, wk_b, kraw, T, HD, HID);
    }
    // w = hidden @ wproj.T    (M=T, N=64, K=7168)
    {
        int waves = (T / 16) * (NH / 64);
        gemm_bf16_nt<<<(waves + 7) / 8, 256, 0, stream>>>(hid_b, wp_b, w_f, T, NH, HID);
    }
    // LN + RoPE on k
    ln_rope_k<<<T, 128, 0, stream>>>(kraw, knw, knb, seq_lens, nb, k_b, T);
    // fused scores -> logits with masking; block = 16 t x 128 s
    {
        int blocks = (T / 16) * (T / 16 / 8);
        logits_kernel<<<blocks, 256, 0, stream>>>(q_b, k_b, w_f,
                                                  seq_lens, nb, logits, T);
    }
    // top-k (values then indices, concatenated in d_out)
    float* outv = (float*)d_out;
    int* outi = (int*)((float*)d_out + (size_t)T * 512);
    topk_kernel<<<T, 256, 0, stream>>>(logits, T, outv, outi);
}
